// ConfSMoE_79285096284557
// MI455X (gfx1250) — matrix operand order repaired and checked by reading, not compiled, not run on hardware
//
#include <hip/hip_runtime.h>
#include <hip/hip_bf16.h>
#include <math.h>

typedef _Float16 v8h  __attribute__((ext_vector_type(8)));
typedef _Float16 v16h __attribute__((ext_vector_type(16)));
typedef float    v8f  __attribute__((ext_vector_type(8)));

#define DMODEL 512
#define DFF    2048
#define NEXP   8
#define EPSLN  1e-5f

__device__ __forceinline__ v8f wmma16x16x32(v16h a, v16h b, v8f c) {
  // D = A(16x32 f16) * B(32x16 f16) + C(16x16 f32)
  return __builtin_amdgcn_wmma_f32_16x16x32_f16(false, a, false, b, (short)0, c,
                                                false, false);
}

__device__ __forceinline__ v16h cat8(v8h lo, v8h hi) {
  v16h r;
#pragma unroll
  for (int i = 0; i < 8; ++i) { r[i] = lo[i]; r[i + 8] = hi[i]; }
  return r;
}

__device__ __forceinline__ float wave_sum(float v) {
#pragma unroll
  for (int off = 16; off > 0; off >>= 1) v += __shfl_xor(v, off, 32);
  return v;
}

// Branchless tanh-gelu: 0.5*v*(1+tanh(u)) == v * sigmoid(2u),
// u = 0.7978845608*(v + 0.044715 v^3).  z = -2u*log2(e); g = v / (1 + 2^z).
__device__ __forceinline__ float gelu_fast(float v) {
  float v2 = v * v;
  float z = -v * (2.3022072f + 0.1029434f * v2);
  float t = __builtin_amdgcn_exp2f(z);
  return v * __builtin_amdgcn_rcpf(1.0f + t);
}

// ---------------------------------------------------------------------------
// Weight conversion: fp32 row-major -> f16 "B-operand" layout:
//   w1t[((e*(H/16)+ht)*16 + n) * D + k]  (K contiguous per output column)
// ---------------------------------------------------------------------------
__global__ void conv_w1_kernel(const float* __restrict__ W1,
                               _Float16* __restrict__ w1t) {
  size_t o = (size_t)blockIdx.x * 256 + threadIdx.x;
  int k = (int)(o & (DMODEL - 1));
  size_t r = o >> 9;                 // / DMODEL
  int n = (int)(r & 15); r >>= 4;
  int ht = (int)(r % (DFF / 16));
  int e  = (int)(r / (DFF / 16));
  size_t in = ((size_t)(e * DMODEL + k)) * DFF + ht * 16 + n;
  w1t[o] = (_Float16)W1[in];
}

__global__ void conv_w2_kernel(const float* __restrict__ W2,
                               _Float16* __restrict__ w2t) {
  size_t o = (size_t)blockIdx.x * 256 + threadIdx.x;
  int k = (int)(o & (DFF - 1));
  size_t r = o >> 11;                // / DFF
  int n = (int)(r & 15); r >>= 4;
  int dt = (int)(r % (DMODEL / 16));
  int e  = (int)(r / (DMODEL / 16));
  size_t in = ((size_t)(e * DFF + k)) * DMODEL + dt * 16 + n;
  w2t[o] = (_Float16)W2[in];
}

// ---------------------------------------------------------------------------
// Router: LN, logits, softmax, top-2, conf; scatter to per-expert lists.
// ---------------------------------------------------------------------------
__global__ void router_kernel(const float* __restrict__ x,
                              const float* __restrict__ Wg,
                              const float* __restrict__ ln_g,
                              const float* __restrict__ ln_b,
                              _Float16* __restrict__ xn_h,
                              float* __restrict__ acc,
                              int* __restrict__ counts,
                              int* __restrict__ lists,
                              float* __restrict__ confs,
                              int ntok) {
  int wave = threadIdx.x >> 5, lane = threadIdx.x & 31;
  int t = blockIdx.x * 8 + wave;
  if (t >= ntok) return;

  const float* xr = x + (size_t)t * DMODEL + lane * 16;
  float v[16];
#pragma unroll
  for (int i = 0; i < 4; ++i) {
    float4 f = ((const float4*)xr)[i];
    v[4 * i + 0] = f.x; v[4 * i + 1] = f.y;
    v[4 * i + 2] = f.z; v[4 * i + 3] = f.w;
  }
  float s = 0.f;
#pragma unroll
  for (int i = 0; i < 16; ++i) s += v[i];
  float mean = wave_sum(s) * (1.0f / DMODEL);
  float q = 0.f;
#pragma unroll
  for (int i = 0; i < 16; ++i) { float d = v[i] - mean; q += d * d; }
  float var = wave_sum(q) * (1.0f / DMODEL);
  float rstd = rsqrtf(var + EPSLN);

  const float* gp = ln_g + lane * 16;
  const float* bp = ln_b + lane * 16;
  float xn[16];
  v16h hv;
#pragma unroll
  for (int i = 0; i < 16; ++i) {
    xn[i] = (v[i] - mean) * rstd * gp[i] + bp[i];
    hv[i] = (_Float16)xn[i];
  }
  *(v16h*)(xn_h + (size_t)t * DMODEL + lane * 16) = hv;

  // residual seed
  float* ar = acc + (size_t)t * DMODEL + lane * 16;
#pragma unroll
  for (int i = 0; i < 4; ++i) ((float4*)ar)[i] = ((const float4*)xr)[i];

  // logits over 8 experts
  float le[NEXP];
#pragma unroll
  for (int e = 0; e < NEXP; ++e) le[e] = 0.f;
#pragma unroll 4
  for (int i = 0; i < 16; ++i) {
    const float4* wr = (const float4*)(Wg + (size_t)(lane * 16 + i) * NEXP);
    float4 w0 = wr[0], w1 = wr[1];
    le[0] += xn[i] * w0.x; le[1] += xn[i] * w0.y;
    le[2] += xn[i] * w0.z; le[3] += xn[i] * w0.w;
    le[4] += xn[i] * w1.x; le[5] += xn[i] * w1.y;
    le[6] += xn[i] * w1.z; le[7] += xn[i] * w1.w;
  }
#pragma unroll
  for (int e = 0; e < NEXP; ++e) le[e] = wave_sum(le[e]);

  if (lane == 0) {
    float mx = le[0];
#pragma unroll
    for (int e = 1; e < NEXP; ++e) mx = fmaxf(mx, le[e]);
    float pe[NEXP];
#pragma unroll
    for (int e = 0; e < NEXP; ++e) pe[e] = __expf(le[e] - mx);
    int i1 = 0; float p1 = pe[0];
#pragma unroll
    for (int e = 1; e < NEXP; ++e) if (pe[e] > p1) { p1 = pe[e]; i1 = e; }
    int i2 = (i1 == 0) ? 1 : 0; float p2 = pe[i2];
#pragma unroll
    for (int e = 0; e < NEXP; ++e)
      if (e != i1 && pe[e] > p2) { p2 = pe[e]; i2 = e; }
    float inv = 1.0f / (p1 + p2);
    int pos1 = atomicAdd(&counts[i1], 1);
    lists[i1 * ntok + pos1] = t; confs[i1 * ntok + pos1] = p1 * inv;
    int pos2 = atomicAdd(&counts[i2], 1);
    lists[i2 * ntok + pos2] = t; confs[i2 * ntok + pos2] = p2 * inv;
  }
}

// ---------------------------------------------------------------------------
// Grouped-GEMM expert FFN. Grid (ntok/16, E), 256 threads = 8 waves.
// Stage:   gathered xn A-tile -> LDS via global_load_async_to_lds_b128.
// Phase 1: h[16,DFF] = gelu(xn_tile @ W1[e] + b1)  -> LDS (f16, XOR-swizzled)
// Phase 2: out = h @ W2[e] + b2, scaled by conf, atomicAdd into acc.
// ---------------------------------------------------------------------------
__global__ void __launch_bounds__(256)
moe_ffn_kernel(const _Float16* __restrict__ xn_h,
               const _Float16* __restrict__ w1t, const float* __restrict__ b1,
               const _Float16* __restrict__ w2t, const float* __restrict__ b2,
               const int* __restrict__ counts, const int* __restrict__ lists,
               const float* __restrict__ confs,
               float* __restrict__ acc, int ntok) {
  __shared__ _Float16 hlds[16 * DFF];     // 64 KB h buffer, 32B-granule XOR swizzle
  __shared__ _Float16 alds[16 * DMODEL];  // 16 KB gathered xn tile, same swizzle

  int e = blockIdx.y;
  int tile = blockIdx.x;
  int cnt = counts[e];
  if (tile * 16 >= cnt) return;

  int wave = threadIdx.x >> 5, lane = threadIdx.x & 31;
  int lm = lane & 15, lh = lane >> 4;

  // Fallback token (row lm) for padded rows
  int idxA = tile * 16 + lm;
  int rowA = lists[e * ntok + ((idxA < cnt) ? idxA : tile * 16)];

  // ---- Stage A-tile: 16 rows x 512 f16 = 1024 x 16B chunks, async to LDS ----
  {
    int tid = threadIdx.x;
#pragma unroll
    for (int p = 0; p < 4; ++p) {
      int c = p * 256 + tid;
      int row = c >> 6;                 // 64 x 16B chunks per row
      int c2 = c & 63;
      int idx = tile * 16 + row;
      int tok = lists[e * ntok + ((idx < cnt) ? idx : tile * 16)];
      unsigned long long ga =
          (unsigned long long)(xn_h + (size_t)tok * DMODEL + c2 * 8);
      int g = c2 >> 1;                  // 32B granule within row
      unsigned la = (unsigned)(unsigned long long)(alds + row * DMODEL) +
                    (unsigned)((((g ^ (row & 7)) << 5) | ((c2 & 1) << 4)));
      asm volatile("global_load_async_to_lds_b128 %0, %1, off"
                   :: "v"(la), "v"(ga) : "memory");
    }
    asm volatile("s_wait_asynccnt 0" ::: "memory");
  }
  __syncthreads();

  // ---- Phase 1 ----
  const _Float16* w1e = w1t + (size_t)e * DMODEL * DFF;
  const _Float16* aRow = alds + lm * DMODEL;
  int swA = lm & 7;
  for (int nt = wave; nt < DFF / 16; nt += 8) {
    v8f c = {};
    const _Float16* bPtr = w1e + ((size_t)(nt * 16 + lm)) * DMODEL + lh * 16;
#pragma unroll 4
    for (int ks = 0; ks < DMODEL / 32; ++ks) {
      // Prefetch this lane's B stream for the next nt tile (one line / 4 ks)
      if ((ks & 3) == 0)
        __builtin_prefetch(bPtr + (size_t)8 * 16 * DMODEL + ks * 32, 0, 3);
      // A from LDS (swizzled); lane-half lh holds K = {lh*16 .. lh*16+15}
      v8h a0 = *(const v8h*)(aRow + (((2 * ks + lh) ^ swA) << 4));
      v8h a1 = *(const v8h*)(aRow + (((2 * ks + lh) ^ swA) << 4) + 8);
      v16h a = cat8(a0, a1);
      v16h b = *(const v16h*)(bPtr + ks * 32);
      c = wmma16x16x32(a, b, c);
    }
    int col = nt * 16 + lm;
    float bias = b1[e * DFF + col];
    int g = col >> 4, w = col & 15;
#pragma unroll
    for (int r = 0; r < 8; ++r) {
      int row = lh * 8 + r;            // C layout: VGPR r, lane-half lh
      float gl = gelu_fast(c[r] + bias);
      hlds[row * DFF + (((g ^ (row & 7)) << 4) | w)] = (_Float16)gl;
    }
  }
  __syncthreads();

  // ---- Phase 2 ----
  const _Float16* w2e = w2t + (size_t)e * DFF * DMODEL;
  v8f zero = {};
  v8f cacc[4];
#pragma unroll
  for (int j = 0; j < 4; ++j) cacc[j] = zero;

  const _Float16* hRow = hlds + lm * DFF;   // A row = lm
  int sw = lm & 7;
  const _Float16* bw2 = w2e + ((size_t)(wave * 4 * 16 + lm)) * DFF + lh * 16;
#pragma unroll 4
  for (int ks = 0; ks < DFF / 32; ++ks) {
    v8h a0 = *(const v8h*)(hRow + (((2 * ks + lh) ^ sw) << 4));
    v8h a1 = *(const v8h*)(hRow + (((2 * ks + lh) ^ sw) << 4) + 8);
    v16h a = cat8(a0, a1);
#pragma unroll
    for (int j = 0; j < 4; ++j) {
      const _Float16* bp = bw2 + (size_t)j * 16 * DFF + ks * 32;
      if ((ks & 3) == 0) __builtin_prefetch(bp + 8 * 32, 0, 3);
      v16h b = *(const v16h*)bp;
      cacc[j] = wmma16x16x32(a, b, cacc[j]);
    }
  }

  int toks[8]; float cfs[8];
#pragma unroll
  for (int r = 0; r < 8; ++r) {
    int row = lh * 8 + r;
    int idx = tile * 16 + row;
    if (idx < cnt) { toks[r] = lists[e * ntok + idx]; cfs[r] = confs[e * ntok + idx]; }
    else           { toks[r] = rowA;                  cfs[r] = 0.f; }
  }
#pragma unroll
  for (int j = 0; j < 4; ++j) {
    int col = (wave * 4 + j) * 16 + lm;
    float bias = b2[e * DMODEL + col];
#pragma unroll
    for (int r = 0; r < 8; ++r) {
      float val = (cacc[j][r] + bias) * cfs[r];
      atomicAdd(acc + (size_t)toks[r] * DMODEL + col, val);
    }
  }
}

// ---------------------------------------------------------------------------
// Final residual LayerNorm: out = LN(acc) with out_g/out_b.
// ---------------------------------------------------------------------------
__global__ void final_ln_kernel(const float* __restrict__ acc,
                                const float* __restrict__ out_g,
                                const float* __restrict__ out_b,
                                float* __restrict__ out, int ntok) {
  int wave = threadIdx.x >> 5, lane = threadIdx.x & 31;
  int t = blockIdx.x * 8 + wave;
  if (t >= ntok) return;
  const float* ar = acc + (size_t)t * DMODEL + lane * 16;
  float v[16];
#pragma unroll
  for (int i = 0; i < 4; ++i) {
    float4 f = ((const float4*)ar)[i];
    v[4 * i + 0] = f.x; v[4 * i + 1] = f.y;
    v[4 * i + 2] = f.z; v[4 * i + 3] = f.w;
  }
  float s = 0.f;
#pragma unroll
  for (int i = 0; i < 16; ++i) s += v[i];
  float mean = wave_sum(s) * (1.0f / DMODEL);
  float q = 0.f;
#pragma unroll
  for (int i = 0; i < 16; ++i) { float d = v[i] - mean; q += d * d; }
  float var = wave_sum(q) * (1.0f / DMODEL);
  float rstd = rsqrtf(var + EPSLN);
  const float* gp = out_g + lane * 16;
  const float* bp = out_b + lane * 16;
  float o[16];
#pragma unroll
  for (int i = 0; i < 16; ++i) o[i] = (v[i] - mean) * rstd * gp[i] + bp[i];
  float* orow = out + (size_t)t * DMODEL + lane * 16;
#pragma unroll
  for (int i = 0; i < 4; ++i) {
    float4 f; f.x = o[4 * i]; f.y = o[4 * i + 1];
    f.z = o[4 * i + 2]; f.w = o[4 * i + 3];
    ((float4*)orow)[i] = f;
  }
}

// ---------------------------------------------------------------------------
extern "C" void kernel_launch(void* const* d_in, const int* in_sizes, int n_in,
                              void* d_out, int out_size, void* d_ws,
                              size_t ws_size, hipStream_t stream) {
  const float* x     = (const float*)d_in[0];
  const float* Wg    = (const float*)d_in[1];
  const float* W1    = (const float*)d_in[2];
  const float* b1    = (const float*)d_in[3];
  const float* W2    = (const float*)d_in[4];
  const float* b2    = (const float*)d_in[5];
  const float* ln_g  = (const float*)d_in[6];
  const float* ln_b  = (const float*)d_in[7];
  const float* out_g = (const float*)d_in[8];
  const float* out_b = (const float*)d_in[9];
  float* out = (float*)d_out;
  (void)n_in; (void)out_size; (void)ws_size;

  const int ntok = in_sizes[0] / DMODEL;   // B*S = 4096

  char* ws = (char*)d_ws;
  size_t off = 0;
  auto carve = [&](size_t bytes) -> char* {
    char* p = ws + off;
    off += (bytes + 255) & ~(size_t)255;
    return p;
  };
  _Float16* xn_h  = (_Float16*)carve((size_t)ntok * DMODEL * sizeof(_Float16));
  _Float16* w1t   = (_Float16*)carve((size_t)NEXP * DMODEL * DFF * sizeof(_Float16));
  _Float16* w2t   = (_Float16*)carve((size_t)NEXP * DFF * DMODEL * sizeof(_Float16));
  float*    accb  = (float*)carve((size_t)ntok * DMODEL * sizeof(float));
  int*      counts= (int*)carve(NEXP * sizeof(int));
  int*      lists = (int*)carve((size_t)NEXP * ntok * sizeof(int));
  float*    confs = (float*)carve((size_t)NEXP * ntok * sizeof(float));

  hipMemsetAsync(counts, 0, NEXP * sizeof(int), stream);

  {
    size_t tot = (size_t)NEXP * DMODEL * DFF;
    conv_w1_kernel<<<dim3((unsigned)(tot / 256)), dim3(256), 0, stream>>>(W1, w1t);
  }
  {
    size_t tot = (size_t)NEXP * DFF * DMODEL;
    conv_w2_kernel<<<dim3((unsigned)(tot / 256)), dim3(256), 0, stream>>>(W2, w2t);
  }
  router_kernel<<<dim3(ntok / 8), dim3(256), 0, stream>>>(
      x, Wg, ln_g, ln_b, xn_h, accb, counts, lists, confs, ntok);
  moe_ffn_kernel<<<dim3(ntok / 16, NEXP), dim3(256), 0, stream>>>(
      xn_h, w1t, b1, w2t, b2, counts, lists, confs, accb, ntok);
  final_ln_kernel<<<dim3(ntok / 8), dim3(256), 0, stream>>>(
      accb, out_g, out_b, out, ntok);
}
